// MixedPrecisionLinear_9990093931081
// MI455X (gfx1250) — compile-verified, matrix operand order checked by reading, not run on hardware
//
#include <hip/hip_runtime.h>
#include <hip/hip_bf16.h>

typedef __attribute__((ext_vector_type(16))) _Float16 v16h;
typedef __attribute__((ext_vector_type(8)))  _Float16 v8h;
typedef __attribute__((ext_vector_type(4)))  _Float16 v4h;
typedef __attribute__((ext_vector_type(8)))  float    v8f;

#define GM 8192   // B*S
#define GN 4096   // OUT
#define GK 4096   // IN

#define TM 128
#define TN 128
#define TK 32
#define LDT 40            // padded LDS row stride in halves (80B)
#define BUFH (TM * LDT)   // halves per LDS buffer per matrix

// ---------------------------------------------------------------------------
// prep_x: fp32 -> fp16 conversion of activations (4 elements / thread)
// ---------------------------------------------------------------------------
__global__ __launch_bounds__(256) void prep_x_kernel(const float* __restrict__ x,
                                                     _Float16* __restrict__ x16) {
    int i = blockIdx.x * blockDim.x + threadIdx.x;   // index in float4 units
    float4 v = ((const float4*)x)[i];
    v4h h;
    h[0] = (_Float16)v.x; h[1] = (_Float16)v.y;
    h[2] = (_Float16)v.z; h[3] = (_Float16)v.w;
    ((v4h*)x16)[i] = h;
}

// ---------------------------------------------------------------------------
// prep_w: fuse 3-tier reconstruction -> fp16 weight (4 elements / thread)
// w = wh + (fp32)wm + (!high && !med) * wl * scale
// (wh and wm are already zero outside their masks per setup)
// ---------------------------------------------------------------------------
__global__ __launch_bounds__(256) void prep_w_kernel(const float* __restrict__ wh,
                                                     const _Float16* __restrict__ wm,
                                                     const int* __restrict__ wl,
                                                     const unsigned char* __restrict__ hm,
                                                     const unsigned char* __restrict__ mm,
                                                     const float* __restrict__ scl,
                                                     _Float16* __restrict__ w16) {
    int i = blockIdx.x * blockDim.x + threadIdx.x;   // index in 4-element units
    float s = scl[0];
    float4 h = ((const float4*)wh)[i];
    v4h   m = ((const v4h*)wm)[i];
    int4  l = ((const int4*)wl)[i];
    uchar4 hb = ((const uchar4*)hm)[i];
    uchar4 mb = ((const uchar4*)mm)[i];

    float r0 = h.x + (float)m[0] + ((!hb.x && !mb.x) ? (float)l.x * s : 0.0f);
    float r1 = h.y + (float)m[1] + ((!hb.y && !mb.y) ? (float)l.y * s : 0.0f);
    float r2 = h.z + (float)m[2] + ((!hb.z && !mb.z) ? (float)l.z * s : 0.0f);
    float r3 = h.w + (float)m[3] + ((!hb.w && !mb.w) ? (float)l.w * s : 0.0f);

    v4h o;
    o[0] = (_Float16)r0; o[1] = (_Float16)r1;
    o[2] = (_Float16)r2; o[3] = (_Float16)r3;
    ((v4h*)w16)[i] = o;
}

// ---------------------------------------------------------------------------
// Tiled WMMA GEMM: C[M,N] = A[M,K] * Bw[N,K]^T + bias
// block = 256 threads (8 waves), tile 128x128, K-step 32
// wave grid 4(M) x 2(N): each wave -> 32x64 region = 2x4 WMMA tiles
// Double-buffered LDS, ONE barrier per K-step. Next-tile global loads are
// issued AFTER the barrier so the barrier fence's loadcnt wait is already
// satisfied (the ds_stores consumed the previous loads); the new loads stay
// outstanding across the whole compute phase and are first waited on at the
// next iteration's ds_stores.
// ---------------------------------------------------------------------------
__global__ __launch_bounds__(256) void gemm_kernel(const _Float16* __restrict__ A,
                                                   const _Float16* __restrict__ Bw,
                                                   const float* __restrict__ bias,
                                                   float* __restrict__ C) {
    __shared__ _Float16 lsA[2 * BUFH];
    __shared__ _Float16 lsB[2 * BUFH];

    const int tid  = threadIdx.x;
    const int lane = tid & 31;
    const int wave = tid >> 5;
    const int wm_  = wave & 3;    // wave row (M)   0..3
    const int wn_  = wave >> 2;   // wave col (N)   0..1

    const int blockN = blockIdx.x * TN;
    const int blockM = blockIdx.y * TM;

    // cooperative global->LDS: thread t loads one 32B half-row
    const int ldRow  = tid >> 1;          // 0..127
    const int ldHalf = (tid & 1) * 16;    // 0 or 16 halves

    const _Float16* gA = A  + (size_t)(blockM + ldRow) * GK + ldHalf;
    const _Float16* gB = Bw + (size_t)(blockN + ldRow) * GK + ldHalf;
    _Float16* sA = lsA + ldRow * LDT + ldHalf;
    _Float16* sB = lsB + ldRow * LDT + ldHalf;

    // fragment addressing: lane l<16 -> row l, K 0..15 ; lane l+16 -> row l, K 16..31
    // (same per-lane K order used for A and B fragments => consistent K mapping)
    const int fr = lane & 15;
    const int fk = (lane >> 4) * 16;
    const _Float16* fA0 = lsA + (wm_ * 32 + fr) * LDT + fk;
    const _Float16* fB0 = lsB + (wn_ * 64 + fr) * LDT + fk;

    v8f zero = {};
    v8f acc[2][4];
#pragma unroll
    for (int i = 0; i < 2; ++i)
#pragma unroll
        for (int j = 0; j < 4; ++j) acc[i][j] = zero;

    // ---- prologue: fetch tile 0 into registers ----
    v8h a0 = *(const v8h*)(gA);
    v8h a1 = *(const v8h*)(gA + 8);
    v8h b0 = *(const v8h*)(gB);
    v8h b1 = *(const v8h*)(gB + 8);
    gA += TK; gB += TK;

    int buf = 0;
    for (int kt = 0; kt < GK; kt += TK) {
        // stage tile kt into LDS[buf]; the staggered loadcnt waits here are
        // the ONLY place the previous global loads are waited on
        const int bo = buf * BUFH;
        *(v8h*)(sA + bo)     = a0;
        *(v8h*)(sA + bo + 8) = a1;
        *(v8h*)(sB + bo)     = b0;
        *(v8h*)(sB + bo + 8) = b1;

        __syncthreads();   // fence: dscnt only (loadcnt already drained above)

        // issue global loads for tile kt+TK AFTER the barrier: they overlap
        // with the fragment ds_loads + 8 WMMAs below
        if (kt + TK < GK) {
            a0 = *(const v8h*)(gA);
            a1 = *(const v8h*)(gA + 8);
            b0 = *(const v8h*)(gB);
            b1 = *(const v8h*)(gB + 8);
            gA += TK; gB += TK;
        }

        v16h af[2], bf[4];
#pragma unroll
        for (int i = 0; i < 2; ++i) {
            const _Float16* p = fA0 + bo + i * 16 * LDT;
            v8h lo = *(const v8h*)(p);
            v8h hi = *(const v8h*)(p + 8);
            af[i] = __builtin_shufflevector(lo, hi, 0, 1, 2, 3, 4, 5, 6, 7,
                                            8, 9, 10, 11, 12, 13, 14, 15);
        }
#pragma unroll
        for (int j = 0; j < 4; ++j) {
            const _Float16* p = fB0 + bo + j * 16 * LDT;
            v8h lo = *(const v8h*)(p);
            v8h hi = *(const v8h*)(p + 8);
            bf[j] = __builtin_shufflevector(lo, hi, 0, 1, 2, 3, 4, 5, 6, 7,
                                            8, 9, 10, 11, 12, 13, 14, 15);
        }

#pragma unroll
        for (int i = 0; i < 2; ++i)
#pragma unroll
            for (int j = 0; j < 4; ++j)
                acc[i][j] = __builtin_amdgcn_wmma_f32_16x16x32_f16(
                    /*neg_a=*/false, af[i], /*neg_b=*/false, bf[j],
                    /*c_mod=*/(short)0, acc[i][j],
                    /*reuse_a=*/false, /*reuse_b=*/false);

        buf ^= 1;
    }

    // epilogue: D layout -> VGPR j: lanes 0-15 (M=j, N=lane), lanes 16-31 (M=8+j)
    const int rowHalf = (lane >> 4) * 8;
    const int colBase = blockN + wn_ * 64 + fr;
    const size_t mBase = (size_t)(blockM + wm_ * 32 + rowHalf);

#pragma unroll
    for (int j2 = 0; j2 < 4; ++j2) {
        const int col = colBase + j2 * 16;
        const float bv = bias[col];
#pragma unroll
        for (int i2 = 0; i2 < 2; ++i2) {
#pragma unroll
            for (int r = 0; r < 8; ++r) {
                size_t m = mBase + i2 * 16 + r;
                C[m * GN + col] = acc[i2][j2][r] + bv;
            }
        }
    }
}

// ---------------------------------------------------------------------------
extern "C" void kernel_launch(void* const* d_in, const int* in_sizes, int n_in,
                              void* d_out, int out_size, void* d_ws, size_t ws_size,
                              hipStream_t stream) {
    const float*         x   = (const float*)d_in[0];
    const float*         wh  = (const float*)d_in[1];
    const _Float16*      wmd = (const _Float16*)d_in[2];
    const int*           wl  = (const int*)d_in[3];
    const unsigned char* hm  = (const unsigned char*)d_in[4];
    const unsigned char* mm  = (const unsigned char*)d_in[5];
    const float*         scl = (const float*)d_in[6];
    const float*         bia = (const float*)d_in[7];
    float*               out = (float*)d_out;

    _Float16* x16 = (_Float16*)d_ws;                                   // 64 MiB
    _Float16* w16 = (_Float16*)((char*)d_ws + (size_t)GM * GK * 2);    // 32 MiB

    // x: 8192*4096 fp32 -> fp16, 4 elems/thread
    prep_x_kernel<<<(GM * GK) / (256 * 4), 256, 0, stream>>>(x, x16);
    // w: 4096*4096 reconstruct -> fp16, 4 elems/thread
    prep_w_kernel<<<(GN * GK) / (256 * 4), 256, 0, stream>>>(wh, wmd, wl, hm, mm, scl, w16);
    // GEMM: grid = (N/128, M/128)
    gemm_kernel<<<dim3(GN / TN, GM / TM, 1), 256, 0, stream>>>(x16, w16, bia, out);
    (void)in_sizes; (void)n_in; (void)out_size; (void)ws_size;
}
